// TransformerCritic_66709432041567
// MI455X (gfx1250) — matrix operand order repaired and checked by reading, not compiled, not run on hardware
//
#include <hip/hip_runtime.h>
#include <hip/hip_bf16.h>
#include <math.h>

typedef __attribute__((ext_vector_type(16))) _Float16 v16h;
typedef __attribute__((ext_vector_type(8)))  float    v8f;
typedef __attribute__((ext_vector_type(4)))  int      v4i;

#define B_   128
#define N_   32
#define OBS_ 128
#define A_   16
#define H_   4
#define DM_  128
#define E_   32
#define FIN_ 16
#define OA_  144
#define OAPK 160   // OA padded to multiple of 32 for WMMA K-steps

#if defined(__gfx1250__) && __has_builtin(__builtin_amdgcn_global_load_async_to_lds_b128)
#define HAVE_ASYNC_LDS 1
#else
#define HAVE_ASYNC_LDS 0
#endif

#if HAVE_ASYNC_LDS
__device__ __forceinline__ void async_cp16(const void* g, void* l) {
  __builtin_amdgcn_global_load_async_to_lds_b128(
      (__attribute__((address_space(1))) v4i*)g,
      (__attribute__((address_space(3))) v4i*)l, 0, 0);
}
__device__ __forceinline__ void async_wait0() {
#if __has_builtin(__builtin_amdgcn_s_wait_asynccnt)
  __builtin_amdgcn_s_wait_asynccnt(0);
#else
  asm volatile("s_wait_asynccnt 0x0" ::: "memory");
#endif
}
#endif

__device__ __forceinline__ float leakyf(float x) { return x >= 0.f ? x : 0.01f * x; }

// ---- WMMA fragment loaders (wave32, 16x16x32 f16) -------------------------
// A 16x32 (MxK): lane L -> m = L&15, kbase = (L>>4)*8;
// halves 0..7 = K[kbase..kbase+7], halves 8..15 = K[16+kbase .. 16+kbase+7]
__device__ __forceinline__ v16h ldsA(const _Float16* src, int lane, int ld) {
  int m = lane & 15;
  int kb = (lane >> 4) << 3;
  const _Float16* row = src + m * ld;
  v16h a;
#pragma unroll
  for (int i = 0; i < 8; ++i) { a[i] = row[kb + i]; a[8 + i] = row[16 + kb + i]; }
  return a;
}
// B 32x16 (KxN) from LDS: lane L -> n = L&15, k order same as ldsA
__device__ __forceinline__ v16h ldsB(const _Float16* src, int lane, int ld) {
  int n = lane & 15;
  int kb = (lane >> 4) << 3;
  v16h b;
#pragma unroll
  for (int i = 0; i < 8; ++i) { b[i] = src[(kb + i) * ld + n]; b[8 + i] = src[(16 + kb + i) * ld + n]; }
  return b;
}
// B where logical B[k][n] = src[n*ld + k]  (transposed source, e.g. q·k^T)
__device__ __forceinline__ v16h ldsBt(const _Float16* src, int lane, int ld) {
  const _Float16* row = src + (lane & 15) * ld;
  int kb = (lane >> 4) << 3;
  v16h b;
#pragma unroll
  for (int i = 0; i < 8; ++i) { b[i] = row[kb + i]; b[8 + i] = row[16 + kb + i]; }
  return b;
}
// B 32x16 from global f16 [k][n] (row stride ld): lane L -> n = L&15, k order same as ldsA
__device__ __forceinline__ v16h glbBh(const _Float16* src, int lane, int ld) {
  int n = lane & 15;
  int kb = (lane >> 4) << 3;
  v16h b;
#pragma unroll
  for (int i = 0; i < 8; ++i) { b[i] = src[(size_t)(kb + i) * ld + n]; b[8 + i] = src[(size_t)(16 + kb + i) * ld + n]; }
  return b;
}

// GEMM: D[MxNn] = act(A[MxK] * Bg[KxNn] + bias); A f16 LDS, Bg f16 global,
// D f16 to LDS.  K = Kiters*32, no bounds (weights pre-padded).
__device__ __forceinline__ void gemm_ab_h(const _Float16* A, int lda,
    const _Float16* Bg, int ldb, int Kiters,
    _Float16* D, int ldd, const float* bias, int do_leaky,
    int Mtiles, int Ntiles, int wave, int nwaves, int lane)
{
  int total = Mtiles * Ntiles;
  for (int t = wave; t < total; t += nwaves) {
    int mt = t / Ntiles, nt = t % Ntiles;
    v8f acc = {};
    for (int kk = 0; kk < Kiters; ++kk) {
      v16h a = ldsA(A + mt * 16 * lda + kk * 32, lane, lda);
      v16h b = glbBh(Bg + (size_t)kk * 32 * ldb + nt * 16, lane, ldb);
      acc = __builtin_amdgcn_wmma_f32_16x16x32_f16(false, a, false, b,
                                                   (short)0, acc, false, false);
    }
    int n = lane & 15;
    int col = nt * 16 + n;
    float bv = bias ? bias[col] : 0.f;
    int mbase = mt * 16 + ((lane >> 4) << 3);
#pragma unroll
    for (int r = 0; r < 8; ++r) {
      float v = acc[r] + bv;
      if (do_leaky) v = leakyf(v);
      D[(mbase + r) * ldd + col] = (_Float16)v;
    }
  }
}

// ---- Kernel 0: one-shot f32 -> f16 weight/activation prep ------------------
// sap16 padded to [H][160][128] (rows 144..159 zero); oa16 = obs||act padded
// to [B][N][160] (cols 144..159 zero); pol16 = [B][N][16].
__global__ __launch_bounds__(256) void k0_prep(
    const float* __restrict__ W_se, const float* __restrict__ W_k,
    const float* __restrict__ W_q,  const float* __restrict__ W_sap,
    const float* __restrict__ W_av, const float* __restrict__ W_f1,
    const float* __restrict__ W_f2,
    const float* __restrict__ states, const float* __restrict__ actions,
    const float* __restrict__ policies,
    _Float16* __restrict__ se16, _Float16* __restrict__ k16,
    _Float16* __restrict__ q16,  _Float16* __restrict__ sap16,
    _Float16* __restrict__ av16, _Float16* __restrict__ f116,
    _Float16* __restrict__ f216, _Float16* __restrict__ oa16,
    _Float16* __restrict__ pol16)
{
  int tid = blockIdx.x * 256 + threadIdx.x;
  int stride = gridDim.x * 256;
  for (int i = tid; i < H_ * OBS_ * DM_; i += stride) se16[i] = (_Float16)W_se[i];
  for (int i = tid; i < H_ * DM_ * DM_; i += stride) {
    k16[i] = (_Float16)W_k[i];
    q16[i] = (_Float16)W_q[i];
  }
  for (int i = tid; i < H_ * OAPK * DM_; i += stride) {
    int h = i / (OAPK * DM_);
    int r = (i / DM_) % OAPK;
    int c = i % DM_;
    sap16[i] = (_Float16)(r < OA_ ? W_sap[((size_t)h * OA_ + r) * DM_ + c] : 0.f);
  }
  for (int i = tid; i < H_ * DM_ * E_; i += stride) av16[i] = (_Float16)W_av[i];
  for (int i = tid; i < DM_ * 64;      i += stride) f116[i] = (_Float16)W_f1[i];
  for (int i = tid; i < 64 * FIN_;     i += stride) f216[i] = (_Float16)W_f2[i];
  for (int i = tid; i < B_ * N_ * OAPK; i += stride) {
    int bn = i / OAPK;
    int c  = i % OAPK;
    float v;
    if (c < OBS_)     v = states[(size_t)bn * OBS_ + c];
    else if (c < OA_) v = actions[(size_t)bn * A_ + (c - OBS_)];
    else              v = 0.f;
    oa16[i] = (_Float16)v;
  }
  for (int i = tid; i < B_ * N_ * A_; i += stride) pol16[i] = (_Float16)policies[i];
}

// ---- Kernel 1: se -> k,q -> softmax(q.k^T/sqrt(DM)) per (b,h) -------------
__global__ __launch_bounds__(256) void k1_attn(
    const float* __restrict__ states,
    const _Float16* __restrict__ se16, const float* __restrict__ b_se,
    const _Float16* __restrict__ k16,  const float* __restrict__ b_k,
    const _Float16* __restrict__ q16,  const float* __restrict__ b_q,
    float* __restrict__ out_w, _Float16* __restrict__ wsW)
{
  __shared__ __align__(16) _Float16 Sbuf[N_ * DM_];
  __shared__ __align__(16) _Float16 Ebuf[N_ * DM_];
  __shared__ __align__(16) _Float16 Kbuf[N_ * DM_];
  __shared__ __align__(16) _Float16 Qbuf[N_ * DM_];
  __shared__ float    Sc[N_ * N_];

  int b = blockIdx.x / H_;
  int h = blockIdx.x % H_;
  int tid = threadIdx.x, lane = tid & 31, wave = tid >> 5;

  if (tid == 0) {
    __builtin_prefetch(se16 + (size_t)h * OBS_ * DM_, 0, 1);
    __builtin_prefetch(k16  + (size_t)h * DM_ * DM_, 0, 1);
  }
  const float* sp = states + (size_t)b * N_ * OBS_;
  for (int i = tid; i < N_ * OBS_; i += 256) Sbuf[i] = (_Float16)sp[i];
  __syncthreads();

  // se = leaky(states @ W_se[h] + b_se[h])
  gemm_ab_h(Sbuf, DM_, se16 + (size_t)h * OBS_ * DM_, DM_, 4,
            Ebuf, DM_, b_se + h * DM_, 1, 2, 8, wave, 8, lane);
  __syncthreads();
  // k = se @ W_k + b_k ; q = se @ W_q + b_q  (independent, no sync between)
  gemm_ab_h(Ebuf, DM_, k16 + (size_t)h * DM_ * DM_, DM_, 4,
            Kbuf, DM_, b_k + h * DM_, 0, 2, 8, wave, 8, lane);
  gemm_ab_h(Ebuf, DM_, q16 + (size_t)h * DM_ * DM_, DM_, 4,
            Qbuf, DM_, b_q + h * DM_, 0, 2, 8, wave, 8, lane);
  __syncthreads();

  // scores = q @ k^T * 1/sqrt(DM)   (2x2 tiles, waves 0..3)
  for (int t = wave; t < 4; t += 8) {
    int mt = t >> 1, nt = t & 1;
    v8f acc = {};
    for (int kk = 0; kk < 4; ++kk) {
      v16h a  = ldsA (Qbuf + mt * 16 * DM_ + kk * 32, lane, DM_);
      v16h bf = ldsBt(Kbuf + nt * 16 * DM_ + kk * 32, lane, DM_);
      acc = __builtin_amdgcn_wmma_f32_16x16x32_f16(false, a, false, bf,
                                                   (short)0, acc, false, false);
    }
    int n = lane & 15;
    int col = nt * 16 + n;
    int mbase = mt * 16 + ((lane >> 4) << 3);
#pragma unroll
    for (int r = 0; r < 8; ++r)
      Sc[(mbase + r) * N_ + col] = acc[r] * 0.08838834764831845f; // 1/sqrt(128)
  }
  __syncthreads();

  // row softmax
  if (tid < N_) {
    float mx = -1e30f;
    for (int c = 0; c < N_; ++c) mx = fmaxf(mx, Sc[tid * N_ + c]);
    float s = 0.f;
    for (int c = 0; c < N_; ++c) {
      float e = __expf(Sc[tid * N_ + c] - mx);
      Sc[tid * N_ + c] = e;
      s += e;
    }
    float inv = 1.f / s;
    for (int c = 0; c < N_; ++c) Sc[tid * N_ + c] *= inv;
  }
  __syncthreads();

  float*    ow = out_w + (((size_t)b * H_ + h) * N_) * N_;
  _Float16* ww = wsW   + (((size_t)b * H_ + h) * N_) * N_;
  for (int i = tid; i < N_ * N_; i += 256) { float v = Sc[i]; ow[i] = v; ww[i] = (_Float16)v; }
}

// ---- Kernel 2: per (b,j): oap -> emb -> av -> node -> MLP -> value --------
__global__ __launch_bounds__(256) void k2_value(
    const _Float16* __restrict__ oa16, const _Float16* __restrict__ pol16,
    const _Float16* __restrict__ sap16, const float* __restrict__ b_sap,
    const _Float16* __restrict__ av16,  const float* __restrict__ b_av,
    const _Float16* __restrict__ f116,  const float* __restrict__ b_f1,
    const _Float16* __restrict__ f216,  const float* __restrict__ b_f2,
    const _Float16* __restrict__ wsW, float* __restrict__ out_v)
{
  __shared__ __align__(16) _Float16 OAP[N_ * OAPK];   // 32 x 160
  __shared__ __align__(16) _Float16 EMB[N_ * DM_];    // 32 x 128
  __shared__ __align__(16) _Float16 AV [N_ * E_];     // 32 x 32
  __shared__ __align__(16) _Float16 Wgt[N_ * N_];     // 32 x 32
  __shared__ __align__(16) _Float16 NODE[N_ * DM_];   // 32 x 128 (H*E concat)
  __shared__ __align__(16) _Float16 H1 [N_ * 64];     // 32 x 64

  int b = blockIdx.x >> 5;
  int j = blockIdx.x & 31;
  int tid = threadIdx.x, lane = tid & 31, wave = tid >> 5;

  if (tid == 0) {
    __builtin_prefetch(sap16, 0, 1);
    __builtin_prefetch(f116, 0, 1);
  }

  // stage oap[b, j, :, :] (pure f16 copy; padded layout prebuilt by k0)
  const _Float16* oa = oa16 + ((size_t)b * N_) * OAPK;
#if HAVE_ASYNC_LDS
  for (int i = tid; i < (N_ * OAPK) / 8; i += 256)      // 640 x 16B chunks
    async_cp16(oa + i * 8, &OAP[i * 8]);
  async_wait0();
#else
  for (int i = tid; i < N_ * OAPK; i += 256) OAP[i] = oa[i];
#endif
  __syncthreads();
  // diagonal row j: replace action tail with policy tail
  if (tid < A_) OAP[j * OAPK + OBS_ + tid] = pol16[((size_t)b * N_ + j) * A_ + tid];

  for (int h = 0; h < H_; ++h) {
    const _Float16* wsrc = wsW + (((size_t)b * H_ + h) * N_) * N_;
#if HAVE_ASYNC_LDS
    if (tid < (N_ * N_) / 8)                            // 128 x 16B chunks
      async_cp16(wsrc + tid * 8, &Wgt[tid * 8]);
    async_wait0();
#else
    for (int i = tid; i < N_ * N_; i += 256) Wgt[i] = wsrc[i];
#endif
    __syncthreads();

    // emb = leaky(oap @ W_sap[h] + b_sap[h])   [32 x 128], K=160 (pre-padded)
    gemm_ab_h(OAP, OAPK, sap16 + (size_t)h * OAPK * DM_, DM_, 5,
              EMB, DM_, b_sap + h * DM_, 1, 2, 8, wave, 8, lane);
    __syncthreads();

    // av = leaky(emb @ W_av[h] + b_av[h])      [32 x 32]
    gemm_ab_h(EMB, DM_, av16 + (size_t)h * DM_ * E_, E_, 4,
              AV, E_, b_av + h * E_, 1, 2, 2, wave, 8, lane);
    __syncthreads();

    // node[:, h*32 : h*32+32] = weights @ av   (K=32, one WMMA per tile)
    for (int t = wave; t < 4; t += 8) {
      int mt = t >> 1, nt = t & 1;
      v16h a  = ldsA(Wgt + mt * 16 * N_, lane, N_);
      v16h bf = ldsB(AV + nt * 16, lane, E_);
      v8f acc = {};
      acc = __builtin_amdgcn_wmma_f32_16x16x32_f16(false, a, false, bf,
                                                   (short)0, acc, false, false);
      int n = lane & 15;
      int col = h * E_ + nt * 16 + n;
      int mbase = mt * 16 + ((lane >> 4) << 3);
#pragma unroll
      for (int r = 0; r < 8; ++r) NODE[(mbase + r) * DM_ + col] = (_Float16)acc[r];
    }
    __syncthreads();
  }

  // h1 = leaky(node @ W_f1 + b_f1)   [32 x 64]
  gemm_ab_h(NODE, DM_, f116, 64, 4, H1, 64, b_f1, 1, 2, 4, wave, 8, lane);
  __syncthreads();

  // value = h1 @ W_f2 + b_f2  [32 x 16] -> d_out[b, i, j, :]
  for (int t = wave; t < 2; t += 8) {
    int mt = t;
    v8f acc = {};
    for (int kk = 0; kk < 2; ++kk) {
      v16h a  = ldsA(H1 + mt * 16 * 64 + kk * 32, lane, 64);
      v16h bf = glbBh(f216 + (size_t)kk * 32 * FIN_, lane, FIN_);
      acc = __builtin_amdgcn_wmma_f32_16x16x32_f16(false, a, false, bf,
                                                   (short)0, acc, false, false);
    }
    int n = lane & 15;
    float bv = b_f2[n];
    int mbase = mt * 16 + ((lane >> 4) << 3);
#pragma unroll
    for (int r = 0; r < 8; ++r) {
      int i = mbase + r;
      out_v[(((size_t)b * N_ + i) * N_ + j) * FIN_ + n] = acc[r] + bv;
    }
  }
}

extern "C" void kernel_launch(void* const* d_in, const int* in_sizes, int n_in,
                              void* d_out, int out_size, void* d_ws, size_t ws_size,
                              hipStream_t stream) {
  const float* states   = (const float*)d_in[0];
  const float* policies = (const float*)d_in[1];
  const float* actions  = (const float*)d_in[2];
  const float* W_se  = (const float*)d_in[3];
  const float* b_se  = (const float*)d_in[4];
  const float* W_k   = (const float*)d_in[5];
  const float* b_k   = (const float*)d_in[6];
  const float* W_q   = (const float*)d_in[7];
  const float* b_q   = (const float*)d_in[8];
  const float* W_sap = (const float*)d_in[9];
  const float* b_sap = (const float*)d_in[10];
  const float* W_av  = (const float*)d_in[11];
  const float* b_av  = (const float*)d_in[12];
  const float* W_f1  = (const float*)d_in[13];
  const float* b_f1  = (const float*)d_in[14];
  const float* W_f2  = (const float*)d_in[15];
  const float* b_f2  = (const float*)d_in[16];

  float* out_v = (float*)d_out;                               // [B,N,N,FIN]
  float* out_w = out_v + (size_t)B_ * N_ * N_ * FIN_;         // [B,H,N,N]

  // f16 workspace layout
  _Float16* p = (_Float16*)d_ws;
  _Float16* wsW   = p;  p += (size_t)B_ * H_ * N_ * N_;   // 524288
  _Float16* se16  = p;  p += (size_t)H_ * OBS_ * DM_;     // 65536
  _Float16* k16   = p;  p += (size_t)H_ * DM_ * DM_;      // 65536
  _Float16* q16   = p;  p += (size_t)H_ * DM_ * DM_;      // 65536
  _Float16* sap16 = p;  p += (size_t)H_ * OAPK * DM_;     // 81920 (padded)
  _Float16* av16  = p;  p += (size_t)H_ * DM_ * E_;       // 16384
  _Float16* f116  = p;  p += (size_t)DM_ * 64;            // 8192
  _Float16* f216  = p;  p += (size_t)64 * FIN_;           // 1024
  _Float16* oa16  = p;  p += (size_t)B_ * N_ * OAPK;      // 655360 (padded)
  _Float16* pol16 = p;  p += (size_t)B_ * N_ * A_;        // 65536

  k0_prep<<<256, 256, 0, stream>>>(W_se, W_k, W_q, W_sap, W_av, W_f1, W_f2,
                                   states, actions, policies,
                                   se16, k16, q16, sap16, av16, f116, f216,
                                   oa16, pol16);
  k1_attn<<<B_ * H_, 256, 0, stream>>>(states, se16, b_se, k16, b_k, q16, b_q,
                                       out_w, wsW);
  k2_value<<<B_ * N_, 256, 0, stream>>>(oa16, pol16, sap16, b_sap, av16, b_av,
                                        f116, b_f1, f216, b_f2,
                                        wsW, out_v);
}
